// Transformer_12532714570565
// MI455X (gfx1250) — matrix operand order repaired and checked by reading, not compiled, not run on hardware
//
#include <hip/hip_runtime.h>

// ---------------------------------------------------------------------------
// Transformer forward for MI455X (gfx1250), wave32, WMMA f16 matmuls.
// All GEMM dims in this model are tile multiples (M%128==0 via grid pad,
// N%64==0, K%32==0) and A is always K-contiguous, so staging is unguarded,
// vectorized, and software-pipelined (global loads for tile t+1 issue before
// the barrier of tile t).
// ---------------------------------------------------------------------------

#define TS   512          // sequence length
#define TDM  512          // model dim D
#define TBA  4            // batch
#define TNH  8            // heads
#define THD  64           // head dim
#define TFF  2048         // ffn hidden
#define TVC  32000        // vocab
#define TMS  (TBA * TS)   // 2048 total rows

typedef __attribute__((ext_vector_type(16))) _Float16 v16h;
typedef __attribute__((ext_vector_type(8)))  _Float16 v8h;
typedef __attribute__((ext_vector_type(8)))  float    v8f;

union Frag { v16h v; v8h h[2]; };

// ------------------------------ generic GEMM -------------------------------
// C[b][m][n] = act( alpha * sum_k A[b][m][k]*B[b][k][n] + bias[n]
//                   + rScale * R[b][m*sRM + n*sRN] )
// bz decomposes as (bz/nhdiv, bz%nhdiv) with separate strides so per-(b,h)
// attention GEMMs share this kernel with plain GEMMs.
// Requirements: sAK==1; BMODE=1 -> sBK==1 (K-contig B), BMODE=0 -> sBN==1.

struct GemmArgs {
  const float* A; long aOut; long aIn; int sAM;
  const float* B; long bOut; long bIn; int sBK; int sBN;
  float*       C; long cOut; long cIn; int sCM;
  const float* bias;
  const float* R; long rOut; long rIn; int sRM; int sRN; float rScale;
  float alpha; int M; int N; int K; int relu; int nhdiv;
};

#define TILE_M 128
#define TILE_N 64
#define TILE_K 32
#define LPITCH (TILE_K + 8)   // 40 halves = 80B: 16B-aligned, conflict-free frags

union UA { float4 f4[4]; float f[16]; };
union UB { float4 f4[2]; float f[8]; };

template <int BMODE>
__global__ __launch_bounds__(256) void gemm_wmma_f16(GemmArgs g) {
  // A stored K-permuted: position half*16+j <-> K = half*8 + (j<8 ? j : j+8)
  // so a lane's 16-half fragment is 32 contiguous bytes.
  __shared__ __align__(16) _Float16 As[TILE_M][LPITCH];
  // B stored transposed: BsT[n][k]; lane fragment = 16 contiguous halves.
  __shared__ __align__(16) _Float16 BsT[TILE_N][LPITCH];

  const int bz = blockIdx.z;
  const int ob = bz / g.nhdiv;
  const int ib = bz - ob * g.nhdiv;
  const float* __restrict__ A = g.A + ob * g.aOut + ib * g.aIn;
  const float* __restrict__ B = g.B + ob * g.bOut + ib * g.bIn;
  float*       __restrict__ C = g.C + ob * g.cOut + ib * g.cIn;
  const float* __restrict__ R = g.R ? (g.R + ob * g.rOut + ib * g.rIn) : (const float*)0;

  const int m0   = blockIdx.y * TILE_M;
  const int n0   = blockIdx.x * TILE_N;
  const int tid  = threadIdx.x;
  const int lane = tid & 31;
  const int wid  = tid >> 5;
  const int wm   = (wid & 3) * 32;   // wave's 32-row slab in tile
  const int wn   = (wid >> 2) * 32;  // wave's 32-col slab in tile
  const int lane15 = lane & 15;
  const int hi     = lane >> 4;      // 0 | 1

  // staging coordinates
  const int sm   = tid >> 1;         // A row in tile, 0..127 (2 threads/row)
  const int half = tid & 1;          // which K-half of the row
  const int sbn  = tid & 63;         // B column in tile
  const int sbk  = (tid >> 6) << 3;  // B k-base: 0,8,16,24

  // per-thread base pointers, advanced by one K-tile per iteration
  const float* pa = A + (m0 + sm) * g.sAM + half * 8;
  const float* pb = (BMODE == 1) ? (B + (n0 + sbn) * g.sBN + sbk)
                                 : (B + sbk * g.sBK + (n0 + sbn));
  const int bStep = (BMODE == 1) ? TILE_K : TILE_K * g.sBK;

  auto loadA = [&](UA& u, const float* p) {
    u.f4[0] = *(const float4*)(p);
    u.f4[1] = *(const float4*)(p + 4);
    u.f4[2] = *(const float4*)(p + 16);
    u.f4[3] = *(const float4*)(p + 20);
  };
  auto loadB = [&](UB& u, const float* p) {
    if (BMODE == 1) {
      u.f4[0] = *(const float4*)(p);
      u.f4[1] = *(const float4*)(p + 4);
    } else {
#pragma unroll
      for (int j = 0; j < 8; ++j) u.f[j] = p[j * g.sBK];
    }
  };

  v8f c00 = {}, c01 = {}, c10 = {}, c11 = {};

  UA ua; UB ub;
  loadA(ua, pa);
  loadB(ub, pb);

  const int ntiles = g.K >> 5;   // K % 32 == 0
  for (int t = 0; t < ntiles; ++t) {
    // ---- convert + store tile t to LDS (consumes ua/ub) ----
    {
      v8h h0, h1;
#pragma unroll
      for (int j = 0; j < 8; ++j) h0[j] = (_Float16)ua.f[j];
#pragma unroll
      for (int j = 0; j < 8; ++j) h1[j] = (_Float16)ua.f[8 + j];
      *(v8h*)&As[sm][half * 16]     = h0;
      *(v8h*)&As[sm][half * 16 + 8] = h1;
      v8h hb;
#pragma unroll
      for (int j = 0; j < 8; ++j) hb[j] = (_Float16)ub.f[j];
      *(v8h*)&BsT[sbn][((sbk & 8) << 1) | ((sbk & 16) >> 1)] = hb;
    }
    // ---- pipeline: issue global loads for tile t+1 before the barrier ----
    if (t + 1 < ntiles) {
      pa += TILE_K;
      pb += bStep;
      loadA(ua, pa);
      loadB(ub, pb);
      __builtin_prefetch(pa + TILE_K, 0, 0);            // t+2 speculative
      __builtin_prefetch(pb + bStep, 0, 0);
    }
    __syncthreads();

    // ---- fragments: 2x ds_load_b128 each ----
    Frag a0, a1, b0, b1;
    const _Float16* ap0 = &As[wm + lane15][hi << 4];
    const _Float16* ap1 = &As[wm + 16 + lane15][hi << 4];
    const _Float16* bp0 = &BsT[wn + lane15][hi << 4];
    const _Float16* bp1 = &BsT[wn + 16 + lane15][hi << 4];
    a0.h[0] = *(const v8h*)(ap0); a0.h[1] = *(const v8h*)(ap0 + 8);
    a1.h[0] = *(const v8h*)(ap1); a1.h[1] = *(const v8h*)(ap1 + 8);
    b0.h[0] = *(const v8h*)(bp0); b0.h[1] = *(const v8h*)(bp0 + 8);
    b1.h[0] = *(const v8h*)(bp1); b1.h[1] = *(const v8h*)(bp1 + 8);

    c00 = __builtin_amdgcn_wmma_f32_16x16x32_f16(false, a0.v, false, b0.v, (short)0, c00, false, false);
    c01 = __builtin_amdgcn_wmma_f32_16x16x32_f16(false, a0.v, false, b1.v, (short)0, c01, false, false);
    c10 = __builtin_amdgcn_wmma_f32_16x16x32_f16(false, a1.v, false, b0.v, (short)0, c10, false, false);
    c11 = __builtin_amdgcn_wmma_f32_16x16x32_f16(false, a1.v, false, b1.v, (short)0, c11, false, false);
    __syncthreads();
  }

  // ---- epilogue: alpha, bias, residual(+scale), relu, store ----
  // C/D layout: VGPR r -> lanes 0-15: M=r, N=lane; lanes 16-31: M=r+8, N=lane-16
  const int row0 = m0 + wm + (hi << 3);
  const int col0 = n0 + wn + lane15;
  auto emit = [&](int row, int col, float acc) {
    float v = g.alpha * acc;
    if (g.bias) v += g.bias[col];
    if (R)      v += g.rScale * R[row * g.sRM + col * g.sRN];
    if (g.relu) v = fmaxf(v, 0.0f);
    C[row * g.sCM + col] = v;
  };
#pragma unroll
  for (int r = 0; r < 8; ++r) {
    emit(row0 + r,      col0,      c00[r]);
    emit(row0 + r,      col0 + 16, c01[r]);
    emit(row0 + 16 + r, col0,      c10[r]);
    emit(row0 + 16 + r, col0 + 16, c11[r]);
  }
}

// ------------------------- LayerNorm (wave per row) ------------------------
__global__ __launch_bounds__(256) void layernorm512(const float* __restrict__ x,
                                                    float* __restrict__ y,
                                                    const float* __restrict__ gamma,
                                                    const float* __restrict__ beta,
                                                    int rows) {
  int row  = blockIdx.x * 8 + (threadIdx.x >> 5);
  int lane = threadIdx.x & 31;
  if (row >= rows) return;
  const float* xr = x + (long)row * TDM;
  float v[16];
  float s = 0.0f;
#pragma unroll
  for (int i = 0; i < 16; ++i) { v[i] = xr[lane + i * 32]; s += v[i]; }
#pragma unroll
  for (int off = 16; off > 0; off >>= 1) s += __shfl_xor(s, off, 32);
  float mean = s * (1.0f / 512.0f);
  float q = 0.0f;
#pragma unroll
  for (int i = 0; i < 16; ++i) { float d = v[i] - mean; q += d * d; }
#pragma unroll
  for (int off = 16; off > 0; off >>= 1) q += __shfl_xor(q, off, 32);
  float inv = rsqrtf(q * (1.0f / 512.0f) + 1e-6f);
  float* yr = y + (long)row * TDM;
#pragma unroll
  for (int i = 0; i < 16; ++i) {
    int d = lane + i * 32;
    yr[d] = (v[i] - mean) * inv * gamma[d] + beta[d];
  }
}

// -------------------------- Softmax (wave per row) -------------------------
__global__ __launch_bounds__(256) void softmax512(float* __restrict__ x, int rows) {
  int row  = blockIdx.x * 8 + (threadIdx.x >> 5);
  int lane = threadIdx.x & 31;
  if (row >= rows) return;
  float* xr = x + (long)row * TS;
  float v[16];
  float mx = -3.4e38f;
#pragma unroll
  for (int i = 0; i < 16; ++i) { v[i] = xr[lane + i * 32]; mx = fmaxf(mx, v[i]); }
#pragma unroll
  for (int off = 16; off > 0; off >>= 1) mx = fmaxf(mx, __shfl_xor(mx, off, 32));
  float s = 0.0f;
#pragma unroll
  for (int i = 0; i < 16; ++i) { v[i] = __expf(v[i] - mx); s += v[i]; }
#pragma unroll
  for (int off = 16; off > 0; off >>= 1) s += __shfl_xor(s, off, 32);
  float inv = 1.0f / s;
#pragma unroll
  for (int i = 0; i < 16; ++i) xr[lane + i * 32] = v[i] * inv;
}

// -------------------- Embedding + positional encoding ----------------------
__global__ __launch_bounds__(256) void embed_pe(const int* __restrict__ tok,
                                                const float* __restrict__ emb,
                                                float* __restrict__ out) {
  long gid = (long)blockIdx.x * blockDim.x + threadIdx.x;
  if (gid >= (long)TMS * TDM) return;
  int row = (int)(gid >> 9);
  int d   = (int)(gid & 511);
  int s   = row & (TS - 1);
  int t   = tok[row];
  float val = emb[(long)t * TDM + d] * 22.62741699796952f;  // sqrt(512)
  float expo = (float)(d & ~1) * (1.0f / 512.0f);
  float ang  = (float)s * powf(10000.0f, -expo);
  val += (d & 1) ? cosf(ang) : sinf(ang);
  out[gid] = val;
}

// ------------------------------- host side ---------------------------------

struct MhaW { const float *bk, *bo, *bq, *bv, *wk, *wo, *wq, *wv; };
struct FfnW { const float *b1, *b2, *w1, *w2; };

extern "C" void kernel_launch(void* const* d_in, const int* in_sizes, int n_in,
                              void* d_out, int out_size, void* d_ws, size_t ws_size,
                              hipStream_t stream) {
  (void)in_sizes; (void)n_in; (void)out_size; (void)ws_size;
  int i = 0;
  auto F = [&](int idx) { return (const float*)d_in[idx]; };
  auto rdM = [&](MhaW& m) {
    m.bk = F(i++); m.bo = F(i++); m.bq = F(i++); m.bv = F(i++);
    m.wk = F(i++); m.wo = F(i++); m.wq = F(i++); m.wv = F(i++);
  };
  auto rdF = [&](FfnW& f) { f.b1 = F(i++); f.b2 = F(i++); f.w1 = F(i++); f.w2 = F(i++); };

  // ---- unpack params (jax pytree: dict keys sorted at every level) ----
  const float* dec_emb = F(i++);
  MhaW dec_cross[4], dec_self[4];
  FfnW dec_ffn[4];
  const float *dec_ln1g[4], *dec_ln1b[4], *dec_ln2g[4], *dec_ln2b[4], *dec_ln3g[4], *dec_ln3b[4];
  for (int l = 0; l < 4; ++l) {
    rdM(dec_cross[l]);
    rdF(dec_ffn[l]);
    dec_ln1b[l] = F(i++); dec_ln1g[l] = F(i++);
    dec_ln2b[l] = F(i++); dec_ln2g[l] = F(i++);
    dec_ln3b[l] = F(i++); dec_ln3g[l] = F(i++);
    rdM(dec_self[l]);
  }
  const float* enc_emb = F(i++);
  MhaW enc_attn[4];
  FfnW enc_ffn[4];
  const float *enc_ln1g[4], *enc_ln1b[4], *enc_ln2g[4], *enc_ln2b[4];
  for (int l = 0; l < 4; ++l) {
    rdM(enc_attn[l]);
    rdF(enc_ffn[l]);
    enc_ln1b[l] = F(i++); enc_ln1g[l] = F(i++);
    enc_ln2b[l] = F(i++); enc_ln2g[l] = F(i++);
  }
  const float* out_b = F(i++);
  const float* out_w = F(i++);
  const int*   seq_input  = (const int*)d_in[i++];
  const float* in_pad     = F(i++);
  const int*   seq_target = (const int*)d_in[i++];
  const float* tgt_pad    = F(i++);
  const float* look_ahead = F(i++);

  // ---- workspace layout ----
  float* p  = (float*)d_ws;
  float* X  = p; p += (long)TMS * TDM;   // activations
  float* T  = p; p += (long)TMS * TDM;   // residual-sum temp
  float* Qb = p; p += (long)TMS * TDM;
  float* Kb = p; p += (long)TMS * TDM;
  float* Vb = p; p += (long)TMS * TDM;
  float* Zb = p; p += (long)TMS * TDM;
  float* EN = p; p += (long)TMS * TDM;   // encoder output
  float* LG = p;                          // logits [B*NH, S, S] = 8.4M floats
  float* H  = LG;                         // FFN hidden aliases logits region

  auto run = [&](const GemmArgs& g, int batch) {
    dim3 grid((g.N + TILE_N - 1) / TILE_N, (g.M + TILE_M - 1) / TILE_M, batch);
    if (g.sBK == 1) gemm_wmma_f16<1><<<grid, 256, 0, stream>>>(g);
    else            gemm_wmma_f16<0><<<grid, 256, 0, stream>>>(g);
  };
  // plain row-major GEMM: C[M,N] = act(A[M,K] @ W[K,N] + bias + rScale*R)
  auto gemm = [&](float* C, const float* A, const float* W, const float* bias,
                  const float* R, float rScale, int M, int N, int K, int relu) {
    GemmArgs g = {};
    g.A = A; g.sAM = K;
    g.B = W; g.sBK = N; g.sBN = 1;
    g.C = C; g.sCM = N;
    g.bias = bias;
    g.R = R; g.sRM = N; g.sRN = 1; g.rScale = rScale;
    g.alpha = 1.0f; g.M = M; g.N = N; g.K = K; g.relu = relu; g.nhdiv = 1;
    run(g, 1);
  };
  auto ln = [&](const float* x, float* y, const float* gg, const float* bb) {
    layernorm512<<<dim3(TMS / 8), 256, 0, stream>>>(x, y, gg, bb, TMS);
  };

  // MHA: outT = (softmax(Qh Kh^T/8 - 1e9*mask) Vh) @ wo + bo + resid
  // maskPerQ: 1 -> mask[b,q,k] ([B,1,S,S]), 0 -> mask[b,k] ([B,1,1,S])
  auto mha = [&](const MhaW& w, const float* xq, const float* xkv,
                 const float* mask, int maskPerQ, float* outT, const float* resid) {
    gemm(Qb, xq,  w.wq, w.bq, 0, 0.0f, TMS, TDM, TDM, 0);
    gemm(Kb, xkv, w.wk, w.bk, 0, 0.0f, TMS, TDM, TDM, 0);
    gemm(Vb, xkv, w.wv, w.bv, 0, 0.0f, TMS, TDM, TDM, 0);

    GemmArgs lg = {};                              // logits = Qh @ Kh^T
    lg.A = Qb; lg.aOut = (long)TS * TDM; lg.aIn = THD; lg.sAM = TDM;
    lg.B = Kb; lg.bOut = (long)TS * TDM; lg.bIn = THD; lg.sBK = 1; lg.sBN = TDM;
    lg.C = LG; lg.cOut = (long)TNH * TS * TS; lg.cIn = (long)TS * TS; lg.sCM = TS;
    lg.R = mask; lg.rOut = maskPerQ ? (long)TS * TS : (long)TS; lg.rIn = 0;
    lg.sRM = maskPerQ ? TS : 0; lg.sRN = 1; lg.rScale = -1e9f;
    lg.alpha = 0.125f;                             // 1/sqrt(64)
    lg.M = TS; lg.N = TS; lg.K = THD; lg.nhdiv = TNH;
    run(lg, TBA * TNH);

    softmax512<<<dim3(TBA * TNH * TS / 8), 256, 0, stream>>>(LG, TBA * TNH * TS);

    GemmArgs av = {};                              // Zh = P @ Vh
    av.A = LG; av.aOut = (long)TNH * TS * TS; av.aIn = (long)TS * TS; av.sAM = TS;
    av.B = Vb; av.bOut = (long)TS * TDM; av.bIn = THD; av.sBK = TDM; av.sBN = 1;
    av.C = Zb; av.cOut = (long)TS * TDM; av.cIn = THD; av.sCM = TDM;
    av.alpha = 1.0f; av.M = TS; av.N = THD; av.K = TS; av.nhdiv = TNH;
    run(av, TBA * TNH);

    gemm(outT, Zb, w.wo, w.bo, resid, 1.0f, TMS, TDM, TDM, 0);
  };

  const long blk = ((long)TMS * TDM + 255) / 256;

  // ------------------------------- encoder --------------------------------
  embed_pe<<<dim3((unsigned)blk), 256, 0, stream>>>(seq_input, enc_emb, X);
  for (int l = 0; l < 4; ++l) {
    mha(enc_attn[l], X, X, in_pad, 0, T, X);
    ln(T, X, enc_ln1g[l], enc_ln1b[l]);
    gemm(H, X, enc_ffn[l].w1, enc_ffn[l].b1, 0, 0.0f, TMS, TFF, TDM, 1);
    gemm(T, H, enc_ffn[l].w2, enc_ffn[l].b2, X, 1.0f, TMS, TDM, TFF, 0);
    ln(T, X, enc_ln2g[l], enc_ln2b[l]);
  }
  hipMemcpyAsync(EN, X, (size_t)TMS * TDM * sizeof(float),
                 hipMemcpyDeviceToDevice, stream);

  // ------------------------------- decoder --------------------------------
  embed_pe<<<dim3((unsigned)blk), 256, 0, stream>>>(seq_target, dec_emb, X);
  for (int l = 0; l < 4; ++l) {
    mha(dec_self[l], X, X, look_ahead, 1, T, X);
    ln(T, X, dec_ln1g[l], dec_ln1b[l]);
    mha(dec_cross[l], X, EN, tgt_pad, 0, T, X);
    ln(T, X, dec_ln2g[l], dec_ln2b[l]);
    gemm(H, X, dec_ffn[l].w1, dec_ffn[l].b1, 0, 0.0f, TMS, TFF, TDM, 1);
    gemm(T, H, dec_ffn[l].w2, dec_ffn[l].b2, X, 1.0f, TMS, TDM, TFF, 0);
    ln(T, X, dec_ln3g[l], dec_ln3b[l]);
  }

  // --------------------------- vocab projection ---------------------------
  gemm((float*)d_out, X, out_w, out_b, 0, 0.0f, TMS, TVC, TDM, 0);
}
